// RowCrossAttention_51445118272187
// MI455X (gfx1250) — matrix operand order repaired and checked by reading, not compiled, not run on hardware
//
#include <hip/hip_runtime.h>
#include <hip/hip_bf16.h>
#include <math.h>

#define R_  64
#define C_  256
#define B_  4
#define E_  768
#define Q_  256
#define QD_ 512
#define H_  12
#define D_  64
#define PAD 776   // LDS row stride in halves: 388 dwords, ≡4 mod 64 -> conflict-free b128 frags

typedef __attribute__((ext_vector_type(16))) __bf16 v16bf;
typedef __attribute__((ext_vector_type(8)))  __bf16 v8bf;
typedef __attribute__((ext_vector_type(8)))  float  v8f;

// ---------------- small prep kernels ----------------

// xsum[r][n][e] = sum_c x[r,c,n,e]; optionally also mirror x as bf16 (for async staging)
__global__ void k_reduce_x(const float* __restrict__ x, float* __restrict__ xsum,
                           __bf16* __restrict__ xbf) {
    int idx = blockIdx.x * 256 + threadIdx.x;            // 196608
    int e = idx % E_;
    int n = (idx / E_) % B_;
    int r = idx / (E_ * B_);
    size_t base = ((size_t)r * C_) * (B_ * E_) + (size_t)n * E_ + e;
    float s = 0.f;
    if (xbf) {
        for (int c = 0; c < C_; ++c) {
            float v = x[base + (size_t)c * (B_ * E_)];
            s += v;
            xbf[base + (size_t)c * (B_ * E_)] = (__bf16)v;
        }
    } else {
        for (int c = 0; c < C_; ++c) s += x[base + (size_t)c * (B_ * E_)];
    }
    xsum[idx] = s;
}

// WT[e][j] = bf16(W[j][e])   (B-panel: K=e rows, N=j contiguous)
__global__ void k_transpose_bf16(const float* __restrict__ W, __bf16* __restrict__ WT) {
    int idx = blockIdx.x * 256 + threadIdx.x;            // 589824
    int j = idx % E_;
    int e = idx / E_;
    WT[(size_t)e * E_ + j] = (__bf16)W[(size_t)j * E_ + e];
}

// obv[h][o] = sum_d Wo[o][h*64+d] * bv[h*64+d]
__global__ void k_obv(const float* __restrict__ Wo, const float* __restrict__ bv,
                      float* __restrict__ obv) {
    int idx = blockIdx.x * 256 + threadIdx.x;            // 9216
    int o = idx % E_;
    int h = idx / E_;
    float s = 0.f;
    for (int d = 0; d < D_; ++d) s += Wo[(size_t)o * E_ + h * D_ + d] * bv[h * D_ + d];
    obv[idx] = s;
}

// qp[q][n][hd] = scaling * (query[q,n,:] . Wq[hd,:] + bq[hd]),  scaling = (1/8)/16
__global__ void k_qproj(const float* __restrict__ query, const float* __restrict__ Wq,
                        const float* __restrict__ bq, float* __restrict__ qp) {
    int idx = blockIdx.x * 256 + threadIdx.x;            // 786432
    int hd = idx % E_;
    int n  = (idx / E_) % B_;
    int q  = idx / (E_ * B_);
    const float* qrow = query + (size_t)(q * B_ + n) * QD_;
    const float* wrow = Wq + (size_t)hd * QD_;
    float s = bq[hd];
    for (int e = 0; e < QD_; ++e) s += qrow[e] * wrow[e];
    qp[idx] = s * 0.0078125f;
}

// ksum[r][n][hd] = xsum[r,n,:] . Wk[hd,:] + C*bk[hd]
__global__ void k_kproj(const float* __restrict__ xsum, const float* __restrict__ Wk,
                        const float* __restrict__ bk, float* __restrict__ ksum) {
    int idx = blockIdx.x * 256 + threadIdx.x;            // 196608
    int hd = idx % E_;
    int n  = (idx / E_) % B_;
    int r  = idx / (E_ * B_);
    const float* xrow = xsum + (size_t)(r * B_ + n) * E_;
    const float* wrow = Wk + (size_t)hd * E_;
    float s = 256.0f * bk[hd];
    for (int e = 0; e < E_; ++e) s += xrow[e] * wrow[e];
    ksum[idx] = s;
}

// per (h,n): logits(q,r) -> softmax over r -> w[h,n,r] = sum_q probs
__global__ void k_attn(const float* __restrict__ qp, const float* __restrict__ ksum,
                       float* __restrict__ wout) {
    __shared__ float probs[Q_][R_ + 1];                  // 66560 B
    int h = blockIdx.x / B_;
    int n = blockIdx.x % B_;
    int q = threadIdx.x;
    float qv[D_];
    const float* qrow = qp + (size_t)(q * B_ + n) * E_ + h * D_;
    for (int d = 0; d < D_; ++d) qv[d] = qrow[d];
    float mx = -1e30f;
    for (int r = 0; r < R_; ++r) {
        const float* krow = ksum + (size_t)(r * B_ + n) * E_ + h * D_;
        float s = 0.f;
        for (int d = 0; d < D_; ++d) s += qv[d] * krow[d];
        probs[q][r] = s;
        mx = fmaxf(mx, s);
    }
    float sum = 0.f;
    for (int r = 0; r < R_; ++r) {
        float ev = __expf(probs[q][r] - mx);
        probs[q][r] = ev;
        sum += ev;
    }
    float inv = 1.0f / sum;
    for (int r = 0; r < R_; ++r) probs[q][r] *= inv;
    __syncthreads();
    if (q < R_) {
        float s = 0.f;
        for (int qq = 0; qq < Q_; ++qq) s += probs[qq][q];
        wout[(h * B_ + n) * R_ + q] = s;
    }
}

// bias2[r][n][j] = bo[j] + sum_h w[h,n,r] * obv[h][j]
__global__ void k_bias2(const float* __restrict__ wbuf, const float* __restrict__ obv,
                        const float* __restrict__ bo, float* __restrict__ bias2) {
    int idx = blockIdx.x * 256 + threadIdx.x;            // 196608
    int j = idx % E_;
    int n = (idx / E_) % B_;
    int r = idx / (E_ * B_);
    float s = bo[j];
    for (int h = 0; h < H_; ++h) s += wbuf[(h * B_ + n) * R_ + r] * obv[h * E_ + j];
    bias2[idx] = s;
}

// ---------------- WMMA fragment helpers ----------------

// 16-bit A 16x32 layout: lanes 0-15 hold M=lm, K={kk..kk+7, kk+16..kk+23};
// lanes 16-31 hold M=lm, K={kk+8..kk+15, kk+24..kk+31}  -> two 8-half chunks.
__device__ __forceinline__ v16bf load_frag_a(const __bf16* buf, int lm, int g, int kk) {
    const v8bf* p0 = reinterpret_cast<const v8bf*>(buf + lm * PAD + kk + g * 8);
    const v8bf* p1 = reinterpret_cast<const v8bf*>(buf + lm * PAD + kk + 16 + g * 8);
    v8bf lo = *p0, hi = *p1;
    v16bf a;
#pragma unroll
    for (int i = 0; i < 8; ++i) { a[i] = lo[i]; a[i + 8] = hi[i]; }
    return a;
}

// 16-bit B 32x16 layout: lane l holds N = n, K = {kk+g*8..+7, kk+16+g*8..+7} (same as A).
__device__ __forceinline__ v16bf load_frag_b(const __bf16* P, int kk, int g, int n) {
    v16bf b;
#pragma unroll
    for (int i = 0; i < 8; ++i) {
        b[i]     = P[(size_t)(kk + g * 8 + i) * E_ + n];
        b[i + 8] = P[(size_t)(kk + 16 + g * 8 + i) * E_ + n];
    }
    return b;
}

// ---------------- fused double-GEMM core (32-row tile staged in Abuf) ----------------
// out-tile = (w ∘ (A @ WvT)) @ WoT + bias2 ; 8 waves x 96 cols, 2 M-tiles per wave.
__device__ __forceinline__ void fused_core(const __bf16* Abuf, __bf16* Tbuf,
                                           const __bf16* __restrict__ WvT,
                                           const __bf16* __restrict__ WoT,
                                           const float* __restrict__ wbuf,
                                           const float* __restrict__ bias2,
                                           float* __restrict__ out,
                                           int r, int n, int ct, int tid) {
    int lane = tid & 31;
    int wave = tid >> 5;
    int g  = lane >> 4;
    int lm = lane & 15;
    int ncolBase = wave * 96;

    v8f acc[2][6];
#pragma unroll
    for (int mt = 0; mt < 2; ++mt)
#pragma unroll
        for (int t = 0; t < 6; ++t)
#pragma unroll
            for (int i = 0; i < 8; ++i) acc[mt][t][i] = 0.f;

    // GEMM 1: tmp = A @ WvT  (stage 6 B frags in distinct regs, reuse each for 2 WMMAs)
    for (int kk = 0; kk < E_; kk += 32) {
        v16bf a0 = load_frag_a(Abuf, lm, g, kk);
        v16bf a1 = load_frag_a(Abuf + 16 * PAD, lm, g, kk);
        const __bf16* brow = WvT + (size_t)(kk + lane) * E_;
        if (kk + 32 < E_)
            __builtin_prefetch(WvT + (size_t)(kk + 32 + lane) * E_ + ncolBase, 0, 1);
        v16bf b[6];
#pragma unroll
        for (int t = 0; t < 6; ++t) b[t] = load_frag_b(WvT, kk, g, ncolBase + t * 16 + lm);
#pragma unroll
        for (int t = 0; t < 6; ++t) {
            acc[0][t] = __builtin_amdgcn_wmma_f32_16x16x32_bf16(
                false, a0, false, b[t], (short)0, acc[0][t], false, false);
            acc[1][t] = __builtin_amdgcn_wmma_f32_16x16x32_bf16(
                false, a1, false, b[t], (short)0, acc[1][t], false, false);
        }
    }

    // head-scale by w[h,n,r], requantize to bf16 into Tbuf
#pragma unroll
    for (int t = 0; t < 6; ++t) {
        int nc = ncolBase + t * 16;
        int h  = nc >> 6;                       // 16-wide tile never crosses a head
        float wh = wbuf[(h * B_ + n) * R_ + r];
#pragma unroll
        for (int mt = 0; mt < 2; ++mt)
#pragma unroll
            for (int i = 0; i < 8; ++i)
                Tbuf[(mt * 16 + g * 8 + i) * PAD + nc + lm] = (__bf16)(wh * acc[mt][t][i]);
    }
    __syncthreads();

    // GEMM 2: out = Tbuf @ WoT
#pragma unroll
    for (int mt = 0; mt < 2; ++mt)
#pragma unroll
        for (int t = 0; t < 6; ++t)
#pragma unroll
            for (int i = 0; i < 8; ++i) acc[mt][t][i] = 0.f;

    for (int kk = 0; kk < E_; kk += 32) {
        v16bf a0 = load_frag_a(Tbuf, lm, g, kk);
        v16bf a1 = load_frag_a(Tbuf + 16 * PAD, lm, g, kk);
        const __bf16* brow = WoT + (size_t)(kk + lane) * E_;
        if (kk + 32 < E_)
            __builtin_prefetch(WoT + (size_t)(kk + 32 + lane) * E_ + ncolBase, 0, 1);
        v16bf b[6];
#pragma unroll
        for (int t = 0; t < 6; ++t) b[t] = load_frag_b(WoT, kk, g, ncolBase + t * 16 + lm);
#pragma unroll
        for (int t = 0; t < 6; ++t) {
            acc[0][t] = __builtin_amdgcn_wmma_f32_16x16x32_bf16(
                false, a0, false, b[t], (short)0, acc[0][t], false, false);
            acc[1][t] = __builtin_amdgcn_wmma_f32_16x16x32_bf16(
                false, a1, false, b[t], (short)0, acc[1][t], false, false);
        }
    }

    // write out + bias2
    float* obase = out + ((size_t)(r * C_ + ct * 32) * B_ + n) * E_;
    const float* b2 = bias2 + (size_t)(r * B_ + n) * E_;
#pragma unroll
    for (int t = 0; t < 6; ++t) {
        int nc = ncolBase + t * 16;
        float bterm = b2[nc + lm];
#pragma unroll
        for (int mt = 0; mt < 2; ++mt)
#pragma unroll
            for (int i = 0; i < 8; ++i) {
                int m = mt * 16 + g * 8 + i;    // D-matrix: VGPR i -> M = g*8+i, N = lm
                obase[(size_t)m * (B_ * E_) + nc + lm] = acc[mt][t][i] + bterm;
            }
    }
}

// Variant A: bf16 mirror of x exists -> stage tile with per-lane ASYNC copies into
// padded LDS, drained by s_wait_asynccnt (CDNA5 async global->LDS path).
__global__ __launch_bounds__(256) void k_fused_async(const __bf16* __restrict__ xbf,
                                                     const __bf16* __restrict__ WvT,
                                                     const __bf16* __restrict__ WoT,
                                                     const float* __restrict__ wbuf,
                                                     const float* __restrict__ bias2,
                                                     float* __restrict__ out) {
    __shared__ __attribute__((aligned(16))) __bf16 Abuf[32 * PAD];
    __shared__ __attribute__((aligned(16))) __bf16 Tbuf[32 * PAD];

    int bid = blockIdx.x;            // 2048 = R * B * (C/32)
    int ct  = bid & 7;
    int n   = (bid >> 3) & 3;
    int r   = bid >> 5;
    int tid = threadIdx.x;

    const __bf16* src = xbf + ((size_t)(r * C_ + ct * 32) * B_ + n) * E_;
    // 32 rows x 1536 B = 3072 chunks of 16 B; 256 lanes x 12 issues
#pragma unroll
    for (int it = 0; it < 12; ++it) {
        int chunk = it * 256 + tid;
        int m = chunk / 96;                       // 96 x 16B chunks per 768-half row
        int o = (chunk % 96) * 8;                 // offset in halves
        unsigned ldsa = (unsigned)(size_t)(Abuf + m * PAD + o);
        const __bf16* gp = src + (size_t)m * (B_ * E_) + o;
        asm volatile("global_load_async_to_lds_b128 %0, %1, off"
                     :: "v"(ldsa), "v"(gp) : "memory");
    }
    asm volatile("s_wait_asynccnt 0x0" ::: "memory");
    __syncthreads();

    fused_core(Abuf, Tbuf, WvT, WoT, wbuf, bias2, out, r, n, ct, tid);
}

// Variant B: fallback when workspace can't hold the bf16 mirror — stage from f32 x.
__global__ __launch_bounds__(256) void k_fused_f32(const float* __restrict__ x,
                                                   const __bf16* __restrict__ WvT,
                                                   const __bf16* __restrict__ WoT,
                                                   const float* __restrict__ wbuf,
                                                   const float* __restrict__ bias2,
                                                   float* __restrict__ out) {
    __shared__ __attribute__((aligned(16))) __bf16 Abuf[32 * PAD];
    __shared__ __attribute__((aligned(16))) __bf16 Tbuf[32 * PAD];

    int bid = blockIdx.x;
    int ct  = bid & 7;
    int n   = (bid >> 3) & 3;
    int r   = bid >> 5;
    int tid = threadIdx.x;

    const float* xbase = x + ((size_t)(r * C_ + ct * 32) * B_ + n) * E_;
    for (int idx = tid; idx < 32 * E_; idx += 256) {
        int m = idx / E_, e = idx % E_;
        Abuf[m * PAD + e] = (__bf16)xbase[(size_t)m * (B_ * E_) + e];
    }
    __syncthreads();

    fused_core(Abuf, Tbuf, WvT, WoT, wbuf, bias2, out, r, n, ct, tid);
}

// ---------------- launch ----------------

extern "C" void kernel_launch(void* const* d_in, const int* in_sizes, int n_in,
                              void* d_out, int out_size, void* d_ws, size_t ws_size,
                              hipStream_t stream) {
    const float* x     = (const float*)d_in[0];
    const float* query = (const float*)d_in[1];
    const float* Wq    = (const float*)d_in[2];
    const float* bq    = (const float*)d_in[3];
    const float* Wk    = (const float*)d_in[4];
    const float* bk    = (const float*)d_in[5];
    const float* Wv    = (const float*)d_in[6];
    const float* bv    = (const float*)d_in[7];
    const float* Wo    = (const float*)d_in[8];
    const float* bo    = (const float*)d_in[9];
    float* out = (float*)d_out;

    char* ws = (char*)d_ws;
    size_t off = 0;
    auto carve = [&](size_t bytes) -> void* {
        void* p = ws + off;
        off = (off + bytes + 255) & ~(size_t)255;
        return p;
    };
    float*  xsum  = (float*)carve((size_t)R_ * B_ * E_ * 4);   // 786 KB
    float*  qp    = (float*)carve((size_t)Q_ * B_ * E_ * 4);   // 3 MB
    float*  ksum  = (float*)carve((size_t)R_ * B_ * E_ * 4);
    float*  wbuf  = (float*)carve((size_t)H_ * B_ * R_ * 4);
    float*  bias2 = (float*)carve((size_t)R_ * B_ * E_ * 4);
    float*  obv   = (float*)carve((size_t)H_ * E_ * 4);
    __bf16* WvT   = (__bf16*)carve((size_t)E_ * E_ * 2);       // 1.18 MB
    __bf16* WoT   = (__bf16*)carve((size_t)E_ * E_ * 2);
    size_t small_off = off;
    __bf16* xbf   = (__bf16*)carve((size_t)R_ * C_ * B_ * E_ * 2);  // 96 MB bf16 mirror
    bool use_async = (ws_size >= off);
    if (!use_async) xbf = nullptr;
    (void)in_sizes; (void)n_in; (void)out_size; (void)small_off;

    k_reduce_x      <<<  768, 256, 0, stream>>>(x, xsum, xbf);
    k_transpose_bf16<<< 2304, 256, 0, stream>>>(Wv, WvT);
    k_transpose_bf16<<< 2304, 256, 0, stream>>>(Wo, WoT);
    k_obv           <<<   36, 256, 0, stream>>>(Wo, bv, obv);
    k_qproj         <<< 3072, 256, 0, stream>>>(query, Wq, bq, qp);
    k_kproj         <<<  768, 256, 0, stream>>>(xsum, Wk, bk, ksum);
    k_attn          <<<   48, 256, 0, stream>>>(qp, ksum, wbuf);
    k_bias2         <<<  768, 256, 0, stream>>>(wbuf, obv, bo, bias2);
    if (use_async)
        k_fused_async<<< 2048, 256, 0, stream>>>(xbf, WvT, WoT, wbuf, bias2, out);
    else
        k_fused_f32  <<< 2048, 256, 0, stream>>>(x, WvT, WoT, wbuf, bias2, out);
}
